// stochastic_module_39024072851463
// MI455X (gfx1250) — matrix operand order repaired and checked by reading, not compiled, not run on hardware
//
#include <hip/hip_runtime.h>
#include <hip/hip_bf16.h>

#define BSZ   65536
#define NDIM  64
#define KMIX  8
#define HDIM  1024
#define LOG2PI 1.8378770664093453f

typedef __bf16 bf16_t;
typedef __attribute__((ext_vector_type(16))) __bf16 v16bf;
typedef __attribute__((ext_vector_type(8)))  __bf16 v8bf;
typedef __attribute__((ext_vector_type(8)))  float  v8f;

#define CAT16(lo, hi) __builtin_shufflevector((lo), (hi), 0,1,2,3,4,5,6,7,8,9,10,11,12,13,14,15)

// LDS layout (dynamic): H tile [64][1032] bf16, B/X staging [256][136] bf16, reduction slot
#define HSTR 1032
#define BSTR 136
#define XSTR 72
#define SMEM_BYTES (64 * HSTR * 2 + 256 * BSTR * 2 + 16)

// CDNA5 async global->LDS copy (ASYNCcnt-tracked, no VGPR data movement).
// vdst = per-lane LDS byte offset, vaddr = 64-bit global address (GV mode).
__device__ __forceinline__ void async_g2l_b128(void* lds_ptr, const void* gptr) {
  unsigned int lds_off = (unsigned int)(unsigned long long)lds_ptr;
  asm volatile("global_load_async_to_lds_b128 %0, %1, off"
               :: "v"(lds_off), "v"(gptr)
               : "memory");
}
__device__ __forceinline__ void wait_asynccnt0() {
  asm volatile("s_wait_asynccnt 0" ::: "memory");
}

// ---------------------------------------------------------------------------
// Prep kernels: bf16 conversion (+ transpose for weights so B-fragments are
// contiguous 16-half runs loadable with 2x b128 per fragment).
// ---------------------------------------------------------------------------
__global__ void cvt_x_kernel(const float* __restrict__ x, bf16_t* __restrict__ o, int n) {
  int i = blockIdx.x * blockDim.x + threadIdx.x;
  if (i < n) o[i] = (bf16_t)x[i];
}

__global__ void cvt_w1t_kernel(const float* __restrict__ W1, bf16_t* __restrict__ o) {
  int i = blockIdx.x * blockDim.x + threadIdx.x;  // 1024*64 threads
  int n = i >> 6, k = i & 63;
  o[(n << 6) | k] = (bf16_t)W1[k * HDIM + n];     // W1T[n][k] = W1[k][n]
}

__global__ void cvt_w2t_kernel(const float* __restrict__ W2, bf16_t* __restrict__ o) {
  int i = blockIdx.x * blockDim.x + threadIdx.x;  // 1024*1024 threads
  int n = i >> 10, k = i & 1023;
  o[(n << 10) | k] = (bf16_t)W2[k * 1024 + n];    // W2T[n][k] = W2[k][n]
}

// ---------------------------------------------------------------------------
// Pi network: log_pi[b, k] = log_softmax(ELU(x @ Wp1 + bp1) @ Wp2 + bp2)
// One wave per sample (wave32), shuffle-broadcast x, shuffle-reduce logits.
// ---------------------------------------------------------------------------
__global__ void pi_kernel(const float* __restrict__ x, const float* __restrict__ Wp1,
                          const float* __restrict__ bp1, const float* __restrict__ Wp2,
                          const float* __restrict__ bp2, float* __restrict__ logpi) {
  const int lane = threadIdx.x & 31;
  const int s = blockIdx.x * 8 + (threadIdx.x >> 5);
  float x0 = x[s * 64 + lane];
  float x1 = x[s * 64 + 32 + lane];
  float h0 = bp1[lane], h1 = bp1[lane + 32];
  for (int kk = 0; kk < 64; ++kk) {
    float xv = (kk < 32) ? __shfl(x0, kk, 32) : __shfl(x1, kk - 32, 32);
    h0 = fmaf(xv, Wp1[kk * 64 + lane], h0);
    h1 = fmaf(xv, Wp1[kk * 64 + 32 + lane], h1);
  }
  h0 = h0 > 0.f ? h0 : (__expf(h0) - 1.f);
  h1 = h1 > 0.f ? h1 : (__expf(h1) - 1.f);
  float lg[8];
#pragma unroll
  for (int k = 0; k < 8; ++k) {
    float p = fmaf(h0, Wp2[lane * 8 + k], h1 * Wp2[(lane + 32) * 8 + k]);
    p += __shfl_xor(p, 1, 32);
    p += __shfl_xor(p, 2, 32);
    p += __shfl_xor(p, 4, 32);
    p += __shfl_xor(p, 8, 32);
    p += __shfl_xor(p, 16, 32);
    lg[k] = p + bp2[k];
  }
  float mx = lg[0];
#pragma unroll
  for (int k = 1; k < 8; ++k) mx = fmaxf(mx, lg[k]);
  float se = 0.f;
#pragma unroll
  for (int k = 0; k < 8; ++k) se += __expf(lg[k] - mx);
  float lse = mx + __logf(se);
  if (lane == 0) {
#pragma unroll
    for (int k = 0; k < 8; ++k) logpi[s * 8 + k] = lg[k] - lse;
  }
}

// ---------------------------------------------------------------------------
// Fused main kernel: GEMM1 (bf16 WMMA) + ELU -> LDS H tile -> GEMM2 (bf16
// WMMA, W2T K-chunks staged through LDS via async loads) -> MDN log-prob +
// logsumexp epilogue.
// Block owns fhatx rows [64*bid, 64*bid+64): those rows supply ALL mixture
// components for output samples b2 = ksrc*8192 + q, q in the tile's q-range.
// ---------------------------------------------------------------------------
__global__ __launch_bounds__(256, 1)
void mdn_main_kernel(const float* __restrict__ y,
                     const float* __restrict__ b1g,
                     const float* __restrict__ b2g,
                     const bf16_t* __restrict__ Xbf,
                     const bf16_t* __restrict__ W1T,
                     const bf16_t* __restrict__ W2T,
                     const float* __restrict__ logpi,
                     float* __restrict__ gAcc) {
  extern __shared__ __align__(16) char smem[];
  bf16_t* sH = (bf16_t*)smem;                                  // [64][HSTR]
  bf16_t* sB = (bf16_t*)(smem + 64 * HSTR * 2);                // [256][BSTR] / X tile
  float*  sRed = (float*)(smem + 64 * HSTR * 2 + 256 * BSTR * 2);

  const int tid = threadIdx.x;
  const int wave = tid >> 5;
  const int lane = tid & 31;
  const int ln = lane & 15;   // C/D: N index; A/B: row/col-in-group
  const int g  = lane >> 4;   // lane group
  const int rb = wave >> 1;   // row band (16 rows each)
  const int cw = wave & 1;    // col half (GEMM1: 512 cols) / col sub (GEMM2: 64 cols)
  const int rowTile = blockIdx.x * 64;
  const int m0 = rb * 16;

  if (tid == 0) sRed[0] = 0.f;

  // ---- Phase 0: stage X tile [64][64] bf16 into LDS (async global->LDS)
  bf16_t* sX = sB;
  {
    const bf16_t* src = Xbf + (size_t)rowTile * 64;
    for (int i = tid; i < 512; i += 256) {
      int r = i >> 3, c = (i & 7) << 3;
      async_g2l_b128(sX + r * XSTR + c, src + r * 64 + c);
    }
    wait_asynccnt0();
  }
  __syncthreads();

  // ---- Phase 1: H = ELU(X @ W1 + b1) -> sH (bf16)
  {
    // A fragments (16x32 bf16): lane(m=ln,g), elem e -> K = 16g + e
    const bf16_t* xb = sX + (m0 + ln) * XSTR;
    v16bf a0 = CAT16(*(const v8bf*)(xb + g * 16),      *(const v8bf*)(xb + 8 + g * 16));
    v16bf a1 = CAT16(*(const v8bf*)(xb + 32 + g * 16), *(const v8bf*)(xb + 40 + g * 16));
#pragma unroll 4
    for (int ct = 0; ct < 32; ++ct) {
      const int n0 = cw * 512 + ct * 16;
      const float bias = b1g[n0 + ln];
      v8f acc;
#pragma unroll
      for (int i = 0; i < 8; ++i) acc[i] = bias;  // fold bias into C
      // B fragments from transposed weights: lane(n=ln,g), elem e -> K = 16g + e
      const bf16_t* bp = W1T + (n0 + ln) * 64;
      v16bf b0 = CAT16(*(const v8bf*)(bp + 16 * g),      *(const v8bf*)(bp + 16 * g + 8));
      v16bf b1f = CAT16(*(const v8bf*)(bp + 32 + 16 * g), *(const v8bf*)(bp + 40 + 16 * g));
      acc = __builtin_amdgcn_wmma_f32_16x16x32_bf16(false, a0, false, b0,  (short)0, acc, false, false);
      acc = __builtin_amdgcn_wmma_f32_16x16x32_bf16(false, a1, false, b1f, (short)0, acc, false, false);
#pragma unroll
      for (int r = 0; r < 8; ++r) {  // D elem r -> row m0 + r + 8g, col n0 + ln
        float h = acc[r];
        h = h > 0.f ? h : (__expf(h) - 1.f);
        sH[(m0 + r + 8 * g) * HSTR + n0 + ln] = (bf16_t)h;
      }
    }
  }

  // ---- Phase 2: fhatx = H @ W2 + b2, fused MDN epilogue per segment pair
#pragma unroll 1
  for (int j = 0; j < 4; ++j) {
    const int ksrc = 2 * j + cw;  // source segment handled by this wave
    v8f accs[8];                  // 4 mu tiles + 4 paired var tiles
#pragma unroll
    for (int ct = 0; ct < 4; ++ct) {
      float bm = b2g[j * 128 + cw * 64 + ct * 16 + ln];
      float bv = b2g[512 + j * 128 + cw * 64 + ct * 16 + ln];
#pragma unroll
      for (int i = 0; i < 8; ++i) { accs[ct][i] = bm; accs[ct + 4][i] = bv; }
    }
#pragma unroll 1
    for (int kb2 = 0; kb2 < 8; ++kb2) {  // K chunks of 128
      __syncthreads();                   // previous consumers done
      // cooperative async stage: 256 W2T rows (128 mu + 128 var cols) x 128 K
#pragma unroll
      for (int idx = tid; idx < 4096; idx += 256) {
        int nb = idx >> 4, ko = (idx & 15) << 3;
        int n = (nb < 128) ? (j * 128 + nb) : (512 + j * 128 + (nb - 128));
        async_g2l_b128(sB + nb * BSTR + ko,
                       W2T + (size_t)n * 1024 + kb2 * 128 + ko);
      }
      wait_asynccnt0();
      __syncthreads();
#pragma unroll
      for (int kb = 0; kb < 4; ++kb) {   // 32-wide WMMA K steps
        const bf16_t* ab = sH + (m0 + ln) * HSTR + kb2 * 128 + kb * 32;
        v16bf a = CAT16(*(const v8bf*)(ab + 16 * g), *(const v8bf*)(ab + 8 + 16 * g));
#pragma unroll
        for (int ct = 0; ct < 8; ++ct) { // 8 independent accumulators -> WMMA pipelining
          int nb = ((ct < 4) ? 0 : 128) + cw * 64 + (ct & 3) * 16 + ln;
          const bf16_t* bb = sB + nb * BSTR + kb * 32 + 16 * g;
          v16bf b = CAT16(*(const v8bf*)bb, *(const v8bf*)(bb + 8));
          accs[ct] = __builtin_amdgcn_wmma_f32_16x16x32_bf16(false, a, false, b, (short)0, accs[ct], false, false);
        }
      }
    }
    // Epilogue: row R = rowTile+m0+r+8g -> k2 = r, q = rowTile/8 + 2rb + g
    const int b2s = ksrc * 8192 + (rowTile >> 3) + 2 * rb + g;
    float sr[8] = {0.f, 0.f, 0.f, 0.f, 0.f, 0.f, 0.f, 0.f};
#pragma unroll
    for (int ct = 0; ct < 4; ++ct) {
      float yv = y[(size_t)b2s * 64 + ct * 16 + ln];  // within-segment col = 16ct + ln
#pragma unroll
      for (int r = 0; r < 8; ++r) {
        float mu = accs[ct][r];
        float var = fmaxf(__expf(accs[ct + 4][r]), 1e-8f);
        float d = yv - mu;
        sr[r] += d * d / var + __logf(var);
      }
    }
    float lp[8];
#pragma unroll
    for (int r = 0; r < 8; ++r) {
      float t = sr[r];  // reduce over the 16 N-lanes of this group
      t += __shfl_xor(t, 1, 32);
      t += __shfl_xor(t, 2, 32);
      t += __shfl_xor(t, 4, 32);
      t += __shfl_xor(t, 8, 32);
      lp[r] = -0.5f * (t + 64.f * LOG2PI) + logpi[b2s * 8 + r];
    }
    float mx = lp[0];
#pragma unroll
    for (int r = 1; r < 8; ++r) mx = fmaxf(mx, lp[r]);
    float se = 0.f;
#pragma unroll
    for (int r = 0; r < 8; ++r) se += __expf(lp[r] - mx);
    float lse = mx + __logf(se);
    if (ln == 0) atomicAdd(sRed, lse);  // ds_add_f32
  }
  __syncthreads();
  if (tid == 0) atomicAdd(gAcc, sRed[0]);
}

__global__ void finalize_kernel(const float* __restrict__ gAcc, float* __restrict__ out) {
  if (threadIdx.x == 0 && blockIdx.x == 0) out[0] = -gAcc[0] * (1.f / (float)BSZ);
}

// ---------------------------------------------------------------------------
extern "C" void kernel_launch(void* const* d_in, const int* in_sizes, int n_in,
                              void* d_out, int out_size, void* d_ws, size_t ws_size,
                              hipStream_t stream) {
  const float* x   = (const float*)d_in[0];
  const float* y   = (const float*)d_in[1];
  const float* W1  = (const float*)d_in[2];
  const float* b1  = (const float*)d_in[3];
  const float* W2  = (const float*)d_in[4];
  const float* b2  = (const float*)d_in[5];
  const float* Wp1 = (const float*)d_in[6];
  const float* bp1 = (const float*)d_in[7];
  const float* Wp2 = (const float*)d_in[8];
  const float* bp2 = (const float*)d_in[9];
  float* out = (float*)d_out;

  // workspace carve (16B aligned): acc | log_pi[B*8] | Xbf | W1T | W2T  (~12.6 MB)
  char* w = (char*)d_ws;
  float*  gAcc  = (float*)w;
  float*  logpi = (float*)(w + 16);
  bf16_t* Xbf   = (bf16_t*)(w + 16 + (size_t)BSZ * KMIX * 4);
  bf16_t* W1T   = Xbf + (size_t)BSZ * NDIM;
  bf16_t* W2T   = W1T + (size_t)HDIM * NDIM;

  hipMemsetAsync(gAcc, 0, 16, stream);
  cvt_x_kernel<<<(BSZ * NDIM) / 256, 256, 0, stream>>>(x, Xbf, BSZ * NDIM);
  cvt_w1t_kernel<<<(HDIM * NDIM) / 256, 256, 0, stream>>>(W1, W1T);
  cvt_w2t_kernel<<<(HDIM * HDIM) / 256, 256, 0, stream>>>(W2, W2T);
  pi_kernel<<<BSZ / 8, 256, 0, stream>>>(x, Wp1, bp1, Wp2, bp2, logpi);
  mdn_main_kernel<<<BSZ / 64, 256, SMEM_BYTES, stream>>>(y, b1, b2, Xbf, W1T, W2T, logpi, gAcc);
  finalize_kernel<<<1, 64, 0, stream>>>(gAcc, out);
}
